// Graphormer_29643864277542
// MI455X (gfx1250) — matrix operand order repaired and checked by reading, not compiled, not run on hardware
//
#include <hip/hip_runtime.h>
#include <hip/hip_bf16.h>

// ---------------------------------------------------------------------------
// Graphormer forward for MI455X (gfx1250, wave32, WMMA bf16)
// ---------------------------------------------------------------------------
// Shapes: B=8, N=256, D=768, H=32, DH=24 (padded to 32), F=3072, L=12, K=9

#define Bsz 8
#define Nn 256
#define Dd 768
#define Hh 32
#define DH 24
#define DHP 32
#define Ff 3072
#define Ll 12
#define Kk 9
#define ROWS (Bsz*Nn)            // 2048 token rows

typedef __attribute__((ext_vector_type(16))) __bf16 v16bf;
typedef __attribute__((ext_vector_type(8)))  float  v8f;
typedef __attribute__((ext_vector_type(4))) unsigned int u32x4;

union Frag { v16bf v; u32x4 q[2]; };

__device__ __forceinline__ unsigned short f2bf(float f) {
  union { float f; unsigned u; } c; c.f = f;
  unsigned r = c.u + 0x7FFFu + ((c.u >> 16) & 1u);   // round-to-nearest-even
  return (unsigned short)(r >> 16);
}
__device__ __forceinline__ float bf2f(unsigned short u) {
  union { unsigned u; float f; } c; c.u = ((unsigned)u) << 16; return c.f;
}
__device__ __forceinline__ int imin(int a, int b){ return a < b ? a : b; }
__device__ __forceinline__ int imax(int a, int b){ return a > b ? a : b; }

// 16-lane-row reductions via DPP row_ror butterflies (no LDS traffic)
__device__ __forceinline__ float rmax16(float x) {
  x = fmaxf(x, __int_as_float(__builtin_amdgcn_update_dpp(0, __float_as_int(x), 0x121, 0xf, 0xf, true)));
  x = fmaxf(x, __int_as_float(__builtin_amdgcn_update_dpp(0, __float_as_int(x), 0x122, 0xf, 0xf, true)));
  x = fmaxf(x, __int_as_float(__builtin_amdgcn_update_dpp(0, __float_as_int(x), 0x124, 0xf, 0xf, true)));
  x = fmaxf(x, __int_as_float(__builtin_amdgcn_update_dpp(0, __float_as_int(x), 0x128, 0xf, 0xf, true)));
  return x;
}
__device__ __forceinline__ float rsum16(float x) {
  x += __int_as_float(__builtin_amdgcn_update_dpp(0, __float_as_int(x), 0x121, 0xf, 0xf, true));
  x += __int_as_float(__builtin_amdgcn_update_dpp(0, __float_as_int(x), 0x122, 0xf, 0xf, true));
  x += __int_as_float(__builtin_amdgcn_update_dpp(0, __float_as_int(x), 0x124, 0xf, 0xf, true));
  x += __int_as_float(__builtin_amdgcn_update_dpp(0, __float_as_int(x), 0x128, 0xf, 0xf, true));
  return x;
}

// ---------------------------------------------------------------------------
// Zero a region (16B granules) — used once to zero padded QKV buffers
// ---------------------------------------------------------------------------
__global__ __launch_bounds__(256)
void zero_kernel(u32x4* __restrict__ p, int n16)
{
  int i = blockIdx.x * 256 + threadIdx.x;
  int stride = gridDim.x * 256;
  for (; i < n16; i += stride) p[i] = (u32x4){0u,0u,0u,0u};
}

// ---------------------------------------------------------------------------
// Weight transpose + f32->bf16 : W[z][K][N] -> Wt[z][N][K]
// ---------------------------------------------------------------------------
__global__ __launch_bounds__(256)
void wtrans_kernel(const float* __restrict__ W, unsigned short* __restrict__ Wt,
                   int K, int N)
{
  __shared__ float tile[32][33];
  int k0 = blockIdx.y * 32, n0 = blockIdx.x * 32;
  const float* Wl = W + (size_t)blockIdx.z * K * N;
  unsigned short* Wtl = Wt + (size_t)blockIdx.z * K * N;
  int t = threadIdx.x;
  int tr = t >> 5, tc = t & 31;          // 8 rows x 32 cols per pass
  #pragma unroll
  for (int i = 0; i < 4; ++i)
    tile[tr + 8 * i][tc] = Wl[(size_t)(k0 + tr + 8 * i) * N + n0 + tc];
  __syncthreads();
  #pragma unroll
  for (int i = 0; i < 4; ++i)
    Wtl[(size_t)(n0 + tr + 8 * i) * K + k0 + tc] = f2bf(tile[tc][tr + 8 * i]);
}

// ---------------------------------------------------------------------------
// Node features: sum of K atom embeddings + in/out degree embeddings
// ---------------------------------------------------------------------------
__global__ __launch_bounds__(256)
void embed_kernel(const int* __restrict__ x, const int* __restrict__ ind,
                  const int* __restrict__ outd, const float* __restrict__ atom,
                  const float* __restrict__ din, const float* __restrict__ dout,
                  float* __restrict__ h)
{
  int row = blockIdx.x;
  int di = imin(imax(ind[row], 0), 5);
  int dp = imin(imax(outd[row], 0), 5);
  int idx[Kk];
  #pragma unroll
  for (int k = 0; k < Kk; ++k) idx[k] = x[row * Kk + k];
  for (int c = threadIdx.x; c < Dd; c += 256) {
    float acc = din[di * Dd + c] + dout[dp * Dd + c];
    #pragma unroll
    for (int k = 0; k < Kk; ++k) acc += atom[(size_t)idx[k] * Dd + c];
    h[(size_t)row * Dd + c] = acc;
  }
}

// ---------------------------------------------------------------------------
// Attention bias: 2*attn_bias + spatial + path  -> [B,H,N,N] bf16
// ---------------------------------------------------------------------------
__global__ __launch_bounds__(256)
void bias_kernel(const float* __restrict__ ab, const int* __restrict__ dist,
                 const float* __restrict__ pdat, const float* __restrict__ semb,
                 const float* __restrict__ pemb_g, unsigned short* __restrict__ bias)
{
  __shared__ float pemb[64 * 7];
  int bi = blockIdx.x;                 // b*N + i
  int b = bi >> 8, i = bi & 255;
  int j = threadIdx.x;
  for (int t = threadIdx.x; t < 64 * 7; t += 256) pemb[t] = pemb_g[t];
  __syncthreads();

  int d = dist[(size_t)bi * Nn + j];
  int spi = imin(imax(d, -1), 2) + 1;
  float denom = (float)imin(imax(d, 1), 2);
  float ab2 = ab[(size_t)bi * Nn + j] * 2.f;
  float pd[14];
  const float* pp = pdat + ((size_t)bi * Nn + j) * 14;
  #pragma unroll
  for (int e = 0; e < 14; ++e) pd[e] = pp[e];

  for (int hh = 0; hh < Hh; ++hh) {
    float pe = 0.f;
    #pragma unroll
    for (int l = 0; l < 2; ++l)
      #pragma unroll
      for (int e = 0; e < 7; ++e)
        pe += pd[l * 7 + e] * pemb[(l * Hh + hh) * 7 + e];
    pe /= denom;
    bias[(((size_t)(b * Hh + hh) * Nn) + i) * Nn + j] =
        f2bf(ab2 + semb[spi * Hh + hh] + pe);
  }
}

// ---------------------------------------------------------------------------
// LayerNorm (D=768) -> bf16 output for WMMA GEMM input
// ---------------------------------------------------------------------------
__global__ __launch_bounds__(256)
void layernorm_bf16(const float* __restrict__ x, const float* __restrict__ sc,
                    const float* __restrict__ bi, unsigned short* __restrict__ y)
{
  __shared__ float red[256];
  int row = blockIdx.x, t = threadIdx.x;
  const float* xr = x + (size_t)row * Dd;
  float v0 = xr[t], v1 = xr[t + 256], v2 = xr[t + 512];
  red[t] = v0 + v1 + v2; __syncthreads();
  for (int o = 128; o > 0; o >>= 1) { if (t < o) red[t] += red[t + o]; __syncthreads(); }
  float mean = red[0] * (1.f / 768.f); __syncthreads();
  float d0 = v0 - mean, d1 = v1 - mean, d2 = v2 - mean;
  red[t] = d0 * d0 + d1 * d1 + d2 * d2; __syncthreads();
  for (int o = 128; o > 0; o >>= 1) { if (t < o) red[t] += red[t + o]; __syncthreads(); }
  float rs = rsqrtf(red[0] * (1.f / 768.f) + 1e-5f);
  unsigned short* yr = y + (size_t)row * Dd;
  yr[t]       = f2bf(d0 * rs * sc[t]       + bi[t]);
  yr[t + 256] = f2bf(d1 * rs * sc[t + 256] + bi[t + 256]);
  yr[t + 512] = f2bf(d2 * rs * sc[t + 512] + bi[t + 512]);
}

// ---------------------------------------------------------------------------
// WMMA bf16 GEMM: C[M,N] = A[M,K](bf16) x Bt[N,K](bf16, pre-transposed)
// 128x128 tile, 8 waves of 64x32 (4x2 WMMA accs), BK=32, double-buffered LDS.
// Compile-time epilogue (no runtime branches):
//   MODE=0: v=(acc+bias); opt GELU; opt +residual; store f32 / bf16 [M,N]
//   MODE=1: v=(acc+bias)*oscale -> bf16 padded [B,H,N,32]   (Q / K)
//   MODE=2: v=(acc+bias)       -> bf16 transposed [B,H,32,N] (V)
// ---------------------------------------------------------------------------
template <int MODE, bool GELU, bool RESID, bool SF32, bool SBF16>
__global__ __launch_bounds__(256)
void gemm_bf16(const unsigned short* __restrict__ A,
               const unsigned short* __restrict__ Bt,
               const float* __restrict__ bias, const float* __restrict__ residual,
               float* __restrict__ Cf, unsigned short* __restrict__ Cbf,
               int M, int N, int K, float oscale)
{
  __shared__ __align__(16) unsigned short As[2][128 * 32];
  __shared__ __align__(16) unsigned short Bs[2][128 * 32];
  int bm = blockIdx.y * 128, bn = blockIdx.x * 128;
  int t = threadIdx.x;
  int wave = t >> 5, lane = t & 31, g = lane >> 4, ln = lane & 15;
  int wm = wave >> 2, wn = wave & 3;

  v8f acc[4][2];
  #pragma unroll
  for (int i = 0; i < 4; ++i)
    #pragma unroll
    for (int j = 0; j < 2; ++j)
      acc[i][j] = (v8f){0.f,0.f,0.f,0.f,0.f,0.f,0.f,0.f};

  int rowS[2], ccS[2];
  #pragma unroll
  for (int i = 0; i < 2; ++i) { int c = t + 256 * i; rowS[i] = c >> 2; ccS[i] = c & 3; }

  u32x4 ra[2], rb[2];
  #pragma unroll
  for (int i = 0; i < 2; ++i) {
    ra[i] = *(const u32x4*)(A  + (size_t)(bm + rowS[i]) * K + 8 * ccS[i]);
    rb[i] = *(const u32x4*)(Bt + (size_t)(bn + rowS[i]) * K + 8 * ccS[i]);
  }
  #pragma unroll
  for (int i = 0; i < 2; ++i) {
    *(u32x4*)(&As[0][rowS[i] * 32 + 8 * ccS[i]]) = ra[i];
    *(u32x4*)(&Bs[0][rowS[i] * 32 + 8 * ccS[i]]) = rb[i];
  }
  __syncthreads();

  int KT = K >> 5;
  for (int kt = 0; kt < KT; ++kt) {
    int cur = kt & 1, nxt = cur ^ 1;
    if (kt + 1 < KT) {                       // issue next-tile global loads early
      int k0 = (kt + 1) * 32;
      #pragma unroll
      for (int i = 0; i < 2; ++i) {
        ra[i] = *(const u32x4*)(A  + (size_t)(bm + rowS[i]) * K + k0 + 8 * ccS[i]);
        rb[i] = *(const u32x4*)(Bt + (size_t)(bn + rowS[i]) * K + k0 + 8 * ccS[i]);
      }
    }
    Frag a[4], bf[2];
    #pragma unroll
    for (int i = 0; i < 4; ++i) {
      int m = 64 * wm + 16 * i + ln;
      a[i].q[0] = *(const u32x4*)(&As[cur][m * 32 + 8 * g]);
      a[i].q[1] = *(const u32x4*)(&As[cur][m * 32 + 16 + 8 * g]);
    }
    #pragma unroll
    for (int j = 0; j < 2; ++j) {
      int n = 32 * wn + 16 * j + ln;
      bf[j].q[0] = *(const u32x4*)(&Bs[cur][n * 32 + 8 * g]);
      bf[j].q[1] = *(const u32x4*)(&Bs[cur][n * 32 + 16 + 8 * g]);
    }
    #pragma unroll
    for (int i = 0; i < 4; ++i)
      #pragma unroll
      for (int j = 0; j < 2; ++j)
        acc[i][j] = __builtin_amdgcn_wmma_f32_16x16x32_bf16(
            false, a[i].v, false, bf[j].v, (short)0, acc[i][j], false, false);
    if (kt + 1 < KT) {
      #pragma unroll
      for (int i = 0; i < 2; ++i) {
        *(u32x4*)(&As[nxt][rowS[i] * 32 + 8 * ccS[i]]) = ra[i];
        *(u32x4*)(&Bs[nxt][rowS[i] * 32 + 8 * ccS[i]]) = rb[i];
      }
    }
    __syncthreads();
  }

  #pragma unroll
  for (int i = 0; i < 4; ++i)
    #pragma unroll
    for (int j = 0; j < 2; ++j) {
      int col = bn + 32 * wn + 16 * j + ln;
      float bsv = bias[col];
      #pragma unroll
      for (int r = 0; r < 8; ++r) {
        int row = bm + 64 * wm + 16 * i + 8 * g + r;
        float v = acc[i][j][r] + bsv;
        if (MODE != 0) {
          v *= oscale;
          int hh = col / DH, d = col - hh * DH;       // d in [0,24)
          int b = row >> 8, n = row & 255;
          size_t dst = (MODE == 1)
              ? ((((size_t)(b * Hh + hh)) * Nn + n) * DHP + d)     // [B,H,N,32]
              : ((((size_t)(b * Hh + hh)) * DHP + d) * Nn + n);    // [B,H,32,N]
          Cbf[dst] = f2bf(v);
        } else {
          if (GELU)  v = 0.5f * v * (1.f + erff(v * 0.70710678118654752f));
          if (RESID) v += residual[(size_t)row * N + col];
          if (SF32)  Cf[(size_t)row * N + col] = v;
          if (SBF16) Cbf[(size_t)row * N + col] = f2bf(v);
        }
      }
    }
}

// ---------------------------------------------------------------------------
// Biased attention, flash-style per wave. One block per (b,h).
// Q,K: [B,H,N,32] bf16 (scale folded in Q); V: [B,H,32,N] bf16 (transposed)
// ---------------------------------------------------------------------------
__global__ __launch_bounds__(256)
void attn_kernel(const unsigned short* __restrict__ qp,
                 const unsigned short* __restrict__ kp,
                 const unsigned short* __restrict__ vt,
                 const unsigned short* __restrict__ bias,
                 unsigned short* __restrict__ outbf)
{
  __shared__ __align__(16) unsigned short Qs[Nn * DHP];
  __shared__ __align__(16) unsigned short Ks[Nn * DHP];
  __shared__ __align__(16) unsigned short Vt[DHP * Nn];
  __shared__ __align__(16) unsigned short Ps[8][16 * 32]; // per-wave P tile
  int bh = blockIdx.x;
  int b = bh >> 5, hh = bh & 31;
  size_t base = (size_t)bh * Nn * DHP;
  int t = threadIdx.x, wave = t >> 5, lane = t & 31, g = lane >> 4, ln = lane & 15;

  #pragma unroll
  for (int i = 0; i < 4; ++i) {
    int c = t + 256 * i;                        // 1024 x 16B each
    ((u32x4*)Qs)[c] = ((const u32x4*)(qp + base))[c];
    ((u32x4*)Ks)[c] = ((const u32x4*)(kp + base))[c];
    ((u32x4*)Vt)[c] = ((const u32x4*)(vt + base))[c];
  }
  __syncthreads();

  const unsigned short* biasbh = bias + (size_t)bh * Nn * Nn;

  for (int mt = wave; mt < 16; mt += 8) {       // 16-row query strips
    Frag aQ;
    int m = 16 * mt + ln;
    aQ.q[0] = *(const u32x4*)(&Qs[m * 32 + 8 * g]);
    aQ.q[1] = *(const u32x4*)(&Qs[m * 32 + 16 + 8 * g]);

    v8f o0 = (v8f){0.f,0.f,0.f,0.f,0.f,0.f,0.f,0.f};
    v8f o1 = o0;
    float mrun[8], lrun[8];
    #pragma unroll
    for (int r = 0; r < 8; ++r) { mrun[r] = -1e30f; lrun[r] = 0.f; }

    for (int ks = 0; ks < 8; ++ks) {            // 32 keys per step
      v8f s[2];
      #pragma unroll
      for (int tt = 0; tt < 2; ++tt) {
        int nt = 2 * ks + tt;
        Frag bK;
        int n = 16 * nt + ln;
        bK.q[0] = *(const u32x4*)(&Ks[n * 32 + 8 * g]);
        bK.q[1] = *(const u32x4*)(&Ks[n * 32 + 16 + 8 * g]);
        v8f z = (v8f){0.f,0.f,0.f,0.f,0.f,0.f,0.f,0.f};
        s[tt] = __builtin_amdgcn_wmma_f32_16x16x32_bf16(
            false, aQ.v, false, bK.v, (short)0, z, false, false);
        #pragma unroll
        for (int r = 0; r < 8; ++r)
          s[tt][r] += bf2f(biasbh[(size_t)(16 * mt + 8 * g + r) * Nn + n]);
      }
      // online softmax update; reductions are DPP row_ror butterflies
      float corr[8];
      #pragma unroll
      for (int r = 0; r < 8; ++r) {
        float tmax = rmax16(fmaxf(s[0][r], s[1][r]));
        float mnew = fmaxf(mrun[r], tmax);
        corr[r] = __expf(mrun[r] - mnew);
        mrun[r] = mnew;
      }
      #pragma unroll
      for (int r = 0; r < 8; ++r) {
        float p0 = __expf(s[0][r] - mrun[r]);
        float p1 = __expf(s[1][r] - mrun[r]);
        s[0][r] = p0; s[1][r] = p1;
        float sum = rsum16(p0 + p1);
        lrun[r] = lrun[r] * corr[r] + sum;
        o0[r] *= corr[r]; o1[r] *= corr[r];
      }
      // C-layout -> A-layout relayout of P through per-wave LDS
      #pragma unroll
      for (int tt = 0; tt < 2; ++tt)
        #pragma unroll
        for (int r = 0; r < 8; ++r)
          Ps[wave][(8 * g + r) * 32 + 16 * tt + ln] = f2bf(s[tt][r]);
      asm volatile("s_wait_dscnt 0" ::: "memory");   // same-wave LDS RAW fence
      Frag aP;
      aP.q[0] = *(const u32x4*)(&Ps[wave][ln * 32 + 8 * g]);
      aP.q[1] = *(const u32x4*)(&Ps[wave][ln * 32 + 16 + 8 * g]);
      Frag bV0, bV1;
      bV0.q[0] = *(const u32x4*)(&Vt[ln * Nn + 32 * ks + 8 * g]);
      bV0.q[1] = *(const u32x4*)(&Vt[ln * Nn + 32 * ks + 16 + 8 * g]);
      bV1.q[0] = *(const u32x4*)(&Vt[(16 + ln) * Nn + 32 * ks + 8 * g]);
      bV1.q[1] = *(const u32x4*)(&Vt[(16 + ln) * Nn + 32 * ks + 16 + 8 * g]);
      o0 = __builtin_amdgcn_wmma_f32_16x16x32_bf16(false, aP.v, false, bV0.v,
                                                   (short)0, o0, false, false);
      o1 = __builtin_amdgcn_wmma_f32_16x16x32_bf16(false, aP.v, false, bV1.v,
                                                   (short)0, o1, false, false);
    }
    // epilogue: normalize and scatter to [B,N,H*24] bf16
    #pragma unroll
    for (int r = 0; r < 8; ++r) {
      int row = 16 * mt + 8 * g + r;
      float inv = 1.f / lrun[r];
      int c0 = ln;
      if (c0 < DH)
        outbf[(size_t)(b * Nn + row) * Dd + hh * DH + c0] = f2bf(o0[r] * inv);
      int c1 = 16 + ln;
      if (c1 < DH)
        outbf[(size_t)(b * Nn + row) * Dd + hh * DH + c1] = f2bf(o1[r] * inv);
    }
  }
}

// ---------------------------------------------------------------------------
// Final LN + projection to 3 -> k_flat (first 6144 floats of d_out)
// ---------------------------------------------------------------------------
__global__ __launch_bounds__(256)
void final_proj(const float* __restrict__ hbuf, const float* __restrict__ sc,
                const float* __restrict__ bi, const float* __restrict__ pjw,
                const float* __restrict__ pjb, float* __restrict__ out3)
{
  __shared__ float red[256];
  int row = blockIdx.x, t = threadIdx.x;
  const float* xr = hbuf + (size_t)row * Dd;
  float v0 = xr[t], v1 = xr[t + 256], v2 = xr[t + 512];
  red[t] = v0 + v1 + v2; __syncthreads();
  for (int o = 128; o > 0; o >>= 1) { if (t < o) red[t] += red[t + o]; __syncthreads(); }
  float mean = red[0] * (1.f / 768.f); __syncthreads();
  float d0 = v0 - mean, d1 = v1 - mean, d2 = v2 - mean;
  red[t] = d0 * d0 + d1 * d1 + d2 * d2; __syncthreads();
  for (int o = 128; o > 0; o >>= 1) { if (t < o) red[t] += red[t + o]; __syncthreads(); }
  float rs = rsqrtf(red[0] * (1.f / 768.f) + 1e-5f);
  float y0 = d0 * rs * sc[t] + bi[t];
  float y1 = d1 * rs * sc[t + 256] + bi[t + 256];
  float y2 = d2 * rs * sc[t + 512] + bi[t + 512];
  float p[3];
  #pragma unroll
  for (int c = 0; c < 3; ++c)
    p[c] = y0 * pjw[t * 3 + c] + y1 * pjw[(t + 256) * 3 + c] + y2 * pjw[(t + 512) * 3 + c];
  for (int c = 0; c < 3; ++c) {
    __syncthreads();
    red[t] = p[c]; __syncthreads();
    for (int o = 128; o > 0; o >>= 1) { if (t < o) red[t] += red[t + o]; __syncthreads(); }
    if (t == 0) out3[(size_t)row * 3 + c] = red[0] + pjb[c];
  }
}

// Gram matrix: gram[b,i,j] = <out3[b,i], out3[b,j]>
__global__ __launch_bounds__(256)
void gram_kernel(const float* __restrict__ k_flat, float* __restrict__ gram)
{
  int bi = blockIdx.x;                  // b*N + i
  int b = bi >> 8;
  int j = threadIdx.x;
  const float* xi = k_flat + (size_t)bi * 3;
  const float* xj = k_flat + (size_t)(b * Nn + j) * 3;
  gram[(size_t)bi * Nn + j] = xi[0] * xj[0] + xi[1] * xj[1] + xi[2] * xj[2];
}

// ---------------------------------------------------------------------------
extern "C" void kernel_launch(void* const* d_in, const int* in_sizes, int n_in,
                              void* d_out, int out_size, void* d_ws, size_t ws_size,
                              hipStream_t stream)
{
  (void)in_sizes; (void)n_in; (void)out_size; (void)ws_size;
  const int*   x          = (const int*)  d_in[0];
  const float* attn_bias  = (const float*)d_in[1];
  const int*   dist       = (const int*)  d_in[2];
  const float* path_data  = (const float*)d_in[3];
  const int*   in_degree  = (const int*)  d_in[4];
  const int*   out_degree = (const int*)  d_in[5];
  const float* atom_emb   = (const float*)d_in[6];
  const float* deg_in     = (const float*)d_in[7];
  const float* deg_out    = (const float*)d_in[8];
  const float* sp_emb     = (const float*)d_in[9];
  const float* pa_emb     = (const float*)d_in[10];
  const float* ln1_s = (const float*)d_in[11];
  const float* ln1_b = (const float*)d_in[12];
  const float* wq = (const float*)d_in[13];
  const float* bq = (const float*)d_in[14];
  const float* wk = (const float*)d_in[15];
  const float* bk = (const float*)d_in[16];
  const float* wv = (const float*)d_in[17];
  const float* bv = (const float*)d_in[18];
  const float* wo = (const float*)d_in[19];
  const float* bo = (const float*)d_in[20];
  const float* ln2_s = (const float*)d_in[21];
  const float* ln2_b = (const float*)d_in[22];
  const float* w1 = (const float*)d_in[23];
  const float* b1 = (const float*)d_in[24];
  const float* w2 = (const float*)d_in[25];
  const float* b2 = (const float*)d_in[26];
  const float* fln_s = (const float*)d_in[27];
  const float* fln_b = (const float*)d_in[28];
  const float* pj_w  = (const float*)d_in[29];
  const float* pj_b  = (const float*)d_in[30];

  char* wsp = (char*)d_ws;
  size_t off = 0;
  auto take = [&](size_t bytes) -> char* {
    char* p = wsp + off;
    off += (bytes + 255) & ~(size_t)255;
    return p;
  };
  float*          h      = (float*)         take((size_t)ROWS * Dd * 4);
  unsigned short* ybf    = (unsigned short*)take((size_t)ROWS * Dd * 2);
  unsigned short* biasbf = (unsigned short*)take((size_t)Bsz * Hh * Nn * Nn * 2);
  unsigned short* qp     = (unsigned short*)take((size_t)Bsz * Hh * Nn * DHP * 2);
  unsigned short* kp     = (unsigned short*)take((size_t)Bsz * Hh * Nn * DHP * 2);
  unsigned short* vtb    = (unsigned short*)take((size_t)Bsz * Hh * DHP * Nn * 2);
  unsigned short* attnbf = (unsigned short*)take((size_t)ROWS * Dd * 2);
  unsigned short* ffnbf  = (unsigned short*)take((size_t)ROWS * Ff * 2);
  unsigned short* wqt    = (unsigned short*)take((size_t)Ll * Dd * Dd * 2);
  unsigned short* wkt    = (unsigned short*)take((size_t)Ll * Dd * Dd * 2);
  unsigned short* wvt    = (unsigned short*)take((size_t)Ll * Dd * Dd * 2);
  unsigned short* wot    = (unsigned short*)take((size_t)Ll * Dd * Dd * 2);
  unsigned short* w1t    = (unsigned short*)take((size_t)Ll * Dd * Ff * 2);
  unsigned short* w2t    = (unsigned short*)take((size_t)Ll * Ff * Dd * 2);

  const float qscale = 0.20412414523193154f;   // 24^-0.5

  // One-time per launch: zero padded QKV buffers (qp,kp,vtb are contiguous)
  zero_kernel<<<2048, 256, 0, stream>>>((u32x4*)qp,
      (int)((size_t)3 * Bsz * Hh * Nn * DHP * 2 / 16));

  // One-time per launch: transpose + convert all weights to bf16 [N,K]
  wtrans_kernel<<<dim3(Dd/32, Dd/32, Ll), 256, 0, stream>>>(wq, wqt, Dd, Dd);
  wtrans_kernel<<<dim3(Dd/32, Dd/32, Ll), 256, 0, stream>>>(wk, wkt, Dd, Dd);
  wtrans_kernel<<<dim3(Dd/32, Dd/32, Ll), 256, 0, stream>>>(wv, wvt, Dd, Dd);
  wtrans_kernel<<<dim3(Dd/32, Dd/32, Ll), 256, 0, stream>>>(wo, wot, Dd, Dd);
  wtrans_kernel<<<dim3(Ff/32, Dd/32, Ll), 256, 0, stream>>>(w1, w1t, Dd, Ff);
  wtrans_kernel<<<dim3(Dd/32, Ff/32, Ll), 256, 0, stream>>>(w2, w2t, Ff, Dd);

  embed_kernel<<<ROWS, 256, 0, stream>>>(x, in_degree, out_degree,
                                         atom_emb, deg_in, deg_out, h);
  bias_kernel<<<ROWS, 256, 0, stream>>>(attn_bias, dist, path_data,
                                        sp_emb, pa_emb, biasbf);

  dim3 gD(Dd / 128, ROWS / 128);   // (6,16)
  dim3 gF(Ff / 128, ROWS / 128);   // (24,16)

  for (int l = 0; l < Ll; ++l) {
    const unsigned short* wqt_l = wqt + (size_t)l * Dd * Dd;
    const unsigned short* wkt_l = wkt + (size_t)l * Dd * Dd;
    const unsigned short* wvt_l = wvt + (size_t)l * Dd * Dd;
    const unsigned short* wot_l = wot + (size_t)l * Dd * Dd;
    const unsigned short* w1t_l = w1t + (size_t)l * Dd * Ff;
    const unsigned short* w2t_l = w2t + (size_t)l * Ff * Dd;

    layernorm_bf16<<<ROWS, 256, 0, stream>>>(h, ln1_s + l * Dd, ln1_b + l * Dd, ybf);
    // Q / K -> padded [B,H,N,32] bf16 (scale folded into Q)
    gemm_bf16<1,false,false,false,true><<<gD, 256, 0, stream>>>(
        ybf, wqt_l, bq + l * Dd, nullptr, nullptr, qp, ROWS, Dd, Dd, qscale);
    gemm_bf16<1,false,false,false,true><<<gD, 256, 0, stream>>>(
        ybf, wkt_l, bk + l * Dd, nullptr, nullptr, kp, ROWS, Dd, Dd, 1.f);
    // V -> transposed [B,H,32,N] bf16
    gemm_bf16<2,false,false,false,true><<<gD, 256, 0, stream>>>(
        ybf, wvt_l, bv + l * Dd, nullptr, nullptr, vtb, ROWS, Dd, Dd, 1.f);
    attn_kernel<<<Bsz * Hh, 256, 0, stream>>>(qp, kp, vtb, biasbf, attnbf);
    // out-proj + residual -> h (f32)
    gemm_bf16<0,false,true,true,false><<<gD, 256, 0, stream>>>(
        attnbf, wot_l, bo + l * Dd, h, h, nullptr, ROWS, Dd, Dd, 1.f);
    layernorm_bf16<<<ROWS, 256, 0, stream>>>(h, ln2_s + l * Dd, ln2_b + l * Dd, ybf);
    // FFN1 + exact GELU -> bf16
    gemm_bf16<0,true,false,false,true><<<gF, 256, 0, stream>>>(
        ybf, w1t_l, b1 + l * Ff, nullptr, nullptr, ffnbf, ROWS, Ff, Dd, 1.f);
    // FFN2 + residual -> h (f32)
    gemm_bf16<0,false,true,true,false><<<gD, 256, 0, stream>>>(
        ffnbf, w2t_l, b2 + l * Dd, h, h, nullptr, ROWS, Dd, Ff, 1.f);
  }

  float* k_flat = (float*)d_out;                 // [2048,3]
  float* gram   = (float*)d_out + (size_t)ROWS * 3;
  final_proj<<<ROWS, 256, 0, stream>>>(h, fln_s, fln_b, pj_w, pj_b, k_flat);
  gram_kernel<<<ROWS, 256, 0, stream>>>(k_flat, gram);
}
